// DifferentiableRBFSVMModel_21474836480611
// MI455X (gfx1250) — compile-verified, matrix operand order checked by reading, not run on hardware
//
#include <hip/hip_runtime.h>
#include <hip/hip_bf16.h>
#include <math.h>

#define GAMMA 0.00390625f   // 1/256
#define D_DIM 256

typedef __attribute__((ext_vector_type(16))) __bf16 v16bf;
typedef __attribute__((ext_vector_type(8)))  float  v8f;

union BeatU  { __bf16 h[8]; uint4 u; };     // one 16-byte fragment beat
union FragU  { uint4 u[2];  v16bf v; };     // two beats -> one WMMA operand

// ---------------------------------------------------------------------------
// Kernel 1: per-SV precompute:  xisq[s] = ||xis_s||^2 ,  coef[s] = alpha*y
// ---------------------------------------------------------------------------
__global__ void rbf_precompute_kernel(const float* __restrict__ xis,
                                      const float* __restrict__ alphas,
                                      const float* __restrict__ yis,
                                      float* __restrict__ xisq,
                                      float* __restrict__ coef) {
    int s = blockIdx.x * blockDim.x + threadIdx.x;
    const float4* p = (const float4*)(xis + (size_t)s * D_DIM);
    float acc = 0.0f;
#pragma unroll 8
    for (int i = 0; i < D_DIM / 4; ++i) {
        float4 v = p[i];
        acc += v.x * v.x + v.y * v.y + v.z * v.z + v.w * v.w;
    }
    xisq[s] = acc;
    coef[s] = alphas[s] * yis[s];
}

// ---------------------------------------------------------------------------
// Kernel 2: fused  (x @ xis^T) -> rbf -> weighted-sum -> sigmoid
//   Block: 128 threads = 4 wave32 (64 x-rows)  ->  N/64 = 256 workgroups.
//   A (x rows) resident in registers as bf16 hi/lo fragments.
//   B (xis tile): double-buffered LDS in WMMA fragment order; each thread
//   stages the NEXT tile's 32 f32 in registers while the current tile's
//   WMMAs run -> global-load latency overlaps the matrix pipe, and only one
//   __syncthreads per tile is needed.
//
//   Fragment beat math (ISA 7.12.2, 16-bit A/B layout): lane L of chunk c
//   holds xis_row[L%16][c*32 + (L/16)*8 + {0..7}]  (beat j=0) and
//         xis_row[L%16][c*32 + 16 + (L/16)*8 + {0..7}]  (beat j=1);
//   two contiguous 8-element groups -> two 16B LDS beats per operand.
// ---------------------------------------------------------------------------
__global__ void __launch_bounds__(128)
rbf_svm_fused_kernel(const float* __restrict__ x,
                     const float* __restrict__ xis,
                     const float* __restrict__ xisq,
                     const float* __restrict__ coef,
                     const float* __restrict__ intercept,
                     float* __restrict__ out,
                     int svs) {
    // [buf][chunk c * 64 + beat j * 32 + lane L]
    __shared__ uint4 ldsHi[2][512];   // 2 x 8 KB
    __shared__ uint4 ldsLo[2][512];   // 2 x 8 KB

    const int tid  = threadIdx.x;
    const int wave = tid >> 5;
    const int lane = tid & 31;
    const int half = lane >> 4;     // 0: C rows M=0..7, 1: M=8..15
    const int l16  = lane & 15;     // N column within 16-wide tile / A row

    const int m0 = blockIdx.x * 64 + wave * 16;   // this wave's 16 x-rows

    // --- A fragments (16 x 256 f32) as bf16 hi/lo, resident for the kernel.
    v16bf a_hi[8], a_lo[8];
    const float* arow = x + (size_t)(m0 + l16) * D_DIM;
#pragma unroll
    for (int c = 0; c < 8; ++c) {
        FragU hi, lo;
#pragma unroll
        for (int j = 0; j < 2; ++j) {
            int kbase = c * 32 + j * 16 + half * 8;
            const float4* g = (const float4*)(arow + kbase);
            float4 f0 = g[0], f1 = g[1];
            float f[8] = {f0.x, f0.y, f0.z, f0.w, f1.x, f1.y, f1.z, f1.w};
            BeatU hb, lb;
#pragma unroll
            for (int e = 0; e < 8; ++e) {
                __bf16 h = (__bf16)f[e];
                hb.h[e] = h;
                lb.h[e] = (__bf16)(f[e] - (float)h);
            }
            hi.u[j] = hb.u;
            lo.u[j] = lb.u;
        }
        a_hi[c] = hi.v;
        a_lo[c] = lo.v;
    }

    // --- ||x_m||^2 for this lane's row, broadcast to the 8 C rows I hold.
    float myxsq = 0.0f;
    {
        const float4* p = (const float4*)arow;
#pragma unroll 8
        for (int i = 0; i < D_DIM / 4; ++i) {
            float4 v = p[i];
            myxsq += v.x * v.x + v.y * v.y + v.z * v.z + v.w * v.w;
        }
    }
    float xsq_m[8];
#pragma unroll
    for (int r = 0; r < 8; ++r)      // C row M = r + 8*half lives in lane (r + 24*half)
        xsq_m[r] = __shfl(myxsq, r + 24 * half, 32);

    float sum_r[8];
#pragma unroll
    for (int r = 0; r < 8; ++r) sum_r[r] = 0.0f;

    // --- Per-thread static beat-group assignment (4 groups of 8 f32 each).
    // group g = tid + gi*128 -> (c = g/64, j = (g/32)&1, L = g&31)
    int    gidx[4];
    size_t goff[4];                  // element offset of the 8-f32 source run
#pragma unroll
    for (int gi = 0; gi < 4; ++gi) {
        int g  = tid + gi * 128;
        int c  = g >> 6;
        int j  = (g >> 5) & 1;
        int L  = g & 31;
        gidx[gi] = g;
        goff[gi] = (size_t)(L & 15) * D_DIM + (c * 32 + j * 16 + (L >> 4) * 8);
    }

    // --- Prologue: stage tile 0 into registers.
    float4 st[4][2];
#pragma unroll
    for (int gi = 0; gi < 4; ++gi) {
        const float4* g4 = (const float4*)(xis + goff[gi]);
        st[gi][0] = g4[0];
        st[gi][1] = g4[1];
    }

    // --- Sweep all support vectors, double-buffered, one barrier per tile.
#pragma unroll 1
    for (int sv0 = 0; sv0 < svs; sv0 += 16) {
        const int buf = (sv0 >> 4) & 1;

        // Convert staged tile -> LDS (bf16 hi/lo, fragment order).
#pragma unroll
        for (int gi = 0; gi < 4; ++gi) {
            float f[8] = {st[gi][0].x, st[gi][0].y, st[gi][0].z, st[gi][0].w,
                          st[gi][1].x, st[gi][1].y, st[gi][1].z, st[gi][1].w};
            BeatU hb, lb;
#pragma unroll
            for (int e = 0; e < 8; ++e) {
                __bf16 h = (__bf16)f[e];
                hb.h[e] = h;
                lb.h[e] = (__bf16)(f[e] - (float)h);
            }
            ldsHi[buf][gidx[gi]] = hb.u;
            ldsLo[buf][gidx[gi]] = lb.u;
        }

        // Stage NEXT tile (loads issue now; waits land in the next iteration,
        // fully overlapped with this tile's WMMA stream).
        if (sv0 + 16 < svs) {
#pragma unroll
            for (int gi = 0; gi < 4; ++gi) {
                const float4* g4 =
                    (const float4*)(xis + (size_t)(sv0 + 16) * D_DIM + goff[gi]);
                st[gi][0] = g4[0];
                st[gi][1] = g4[1];
            }
        }

        __syncthreads();   // buf stores visible; also fences buf^1 reuse

        // 16x16 cross tile via bf16 split WMMA: xh*yh + xh*yl + xl*yh.
        v8f acc = {};
#pragma unroll
        for (int c = 0; c < 8; ++c) {
            FragU bh, bl;
            bh.u[0] = ldsHi[buf][c * 64 + lane];
            bh.u[1] = ldsHi[buf][c * 64 + 32 + lane];
            bl.u[0] = ldsLo[buf][c * 64 + lane];
            bl.u[1] = ldsLo[buf][c * 64 + 32 + lane];
            acc = __builtin_amdgcn_wmma_f32_16x16x32_bf16(
                      false, a_hi[c], false, bh.v, (short)0, acc, false, false);
            acc = __builtin_amdgcn_wmma_f32_16x16x32_bf16(
                      false, a_hi[c], false, bl.v, (short)0, acc, false, false);
            acc = __builtin_amdgcn_wmma_f32_16x16x32_bf16(
                      false, a_lo[c], false, bh.v, (short)0, acc, false, false);
        }

        // Fused RBF epilogue: lane owns column N = sv0 + l16, rows M = r + 8*half.
        float xin = xisq[sv0 + l16];
        float cfn = coef[sv0 + l16];
#pragma unroll
        for (int r = 0; r < 8; ++r) {
            float dist = xsq_m[r] + xin - 2.0f * acc[r];
            dist = fmaxf(dist, 0.0f);
            sum_r[r] += cfn * __expf(-GAMMA * dist);
        }
    }

    // --- Reduce across the 16 N-lanes of each half (butterfly stays in-half).
#pragma unroll
    for (int r = 0; r < 8; ++r) {
#pragma unroll
        for (int m = 8; m >= 1; m >>= 1)
            sum_r[r] += __shfl_xor(sum_r[r], m, 32);
    }

    // --- Intercept + sigmoid, one writer per half (lanes 0 and 16).
    if (l16 == 0) {
        float b = intercept[0];
#pragma unroll
        for (int r = 0; r < 8; ++r) {
            float v = sum_r[r] + b;
            out[m0 + r + 8 * half] = 1.0f / (1.0f + __expf(-v));
        }
    }
}

// ---------------------------------------------------------------------------
// Host launcher
// ---------------------------------------------------------------------------
extern "C" void kernel_launch(void* const* d_in, const int* in_sizes, int n_in,
                              void* d_out, int out_size, void* d_ws, size_t ws_size,
                              hipStream_t stream) {
    const float* x         = (const float*)d_in[0];  // [N, 256]
    const float* alphas    = (const float*)d_in[1];  // [SVs]
    const float* xis       = (const float*)d_in[2];  // [SVs, 256]
    const float* yis       = (const float*)d_in[3];  // [SVs]
    const float* intercept = (const float*)d_in[4];  // [1]
    float* out = (float*)d_out;

    const int N   = in_sizes[0] / D_DIM;   // 16384
    const int SVs = in_sizes[1];           // 8192

    float* xisq = (float*)d_ws;
    float* coef = xisq + SVs;

    rbf_precompute_kernel<<<SVs / 256, 256, 0, stream>>>(xis, alphas, yis, xisq, coef);
    rbf_svm_fused_kernel<<<N / 64, 128, 0, stream>>>(x, xis, xisq, coef, intercept,
                                                     out, SVs);
}
